// Informer_26328149525305
// MI455X (gfx1250) — matrix operand order repaired and checked by reading, not compiled, not run on hardware
//
#include <hip/hip_runtime.h>
#include <math.h>

typedef __attribute__((ext_vector_type(16))) _Float16 v16h;
typedef __attribute__((ext_vector_type(8)))  _Float16 v8h;
typedef __attribute__((ext_vector_type(8)))  float    v8f;

#define DM   512
#define NH   8
#define DHD  64
#define DFF  2048
#define NB   8
#define LN_EPS 1e-5f

// LDS tile geometry: 64x64 tile, rows padded to 72 halves (bank-conflict-free:
// row stride 144B -> bank step 36, gcd(36,64)=4 but all 16 lane rows distinct mod 64)
#define KT   64
#define LDP  72

// ---------------------------------------------------------------- helpers
__device__ __forceinline__ float gelu_f(float x) {
  return 0.5f * x * (1.0f + erff(x * 0.70710678118654752f));
}
__device__ __forceinline__ unsigned wang_hash(unsigned s) {
  s = (s ^ 61u) ^ (s >> 16); s *= 9u; s ^= s >> 4; s *= 0x27d4eb2du; s ^= s >> 15;
  return s;
}

// ---------------------------------------------------------------- WMMA GEMM (LDS-staged f16)
// C[m,n] = ACT( scale * sum_k A[m,k]*B(k,n) + bias[n] )
// TRANSB=1: B stored [N,K] row-major (linear layers: W[N,K], scores: K-matrix)
// TRANSB=0: B stored [K,N] row-major (attn @ V)
// Batched over blockIdx.z: z=(b*Hd+h); per-operand offset b*s?b + h*s?h.
// Requires M%64==0, N%64==0, K%64==0 (guaranteed by all call sites).
template<int TRANSB, int ACT>
__global__ __launch_bounds__(128)
void wmma_gemm(const float* __restrict__ A, int lda, long long sAb, long long sAh,
               const float* __restrict__ Bm, int ldb, long long sBb, long long sBh,
               float* __restrict__ C, int ldc, long long sCb, long long sCh,
               const float* __restrict__ bias,
               int M, int N, int K, int Hd, float scale)
{
  __shared__ __align__(16) _Float16 Ah[64 * LDP];   // Ah[m][k]
  __shared__ __align__(16) _Float16 Bh[64 * LDP];   // Bh[n][k]

  int z  = blockIdx.z;
  int zb = z / Hd, zh = z % Hd;
  A  += zb * sAb + zh * sAh;
  Bm += zb * sBb + zh * sBh;
  C  += zb * sCb + zh * sCh;

  int wave = threadIdx.x >> 5;
  int lane = threadIdx.x & 31;
  int wm = wave >> 1, wn = wave & 1;
  int row0 = blockIdx.y * 64;
  int col0 = blockIdx.x * 64;
  int lm = lane & 15, g = lane >> 4;

  v8f acc00 = {}, acc01 = {}, acc10 = {}, acc11 = {};

  for (int kk = 0; kk < K; kk += KT) {
    // ---- stage A tile (64 rows x 64 k) as f16 into LDS, converted once
#pragma unroll
    for (int i = 0; i < 4; ++i) {
      int u = threadIdx.x + 128 * i;          // 512 units of 8 elements
      int r = u >> 3, c8 = (u & 7) * 8;
      const float* src = A + (long long)(row0 + r) * lda + kk + c8;
      _Float16* dst = &Ah[r * LDP + c8];
#pragma unroll
      for (int j = 0; j < 8; ++j) dst[j] = (_Float16)src[j];
    }
    // ---- stage B tile as Bh[n][k]
    if (TRANSB) {
#pragma unroll
      for (int i = 0; i < 4; ++i) {
        int u = threadIdx.x + 128 * i;
        int r = u >> 3, c8 = (u & 7) * 8;
        const float* src = Bm + (long long)(col0 + r) * ldb + kk + c8;
        _Float16* dst = &Bh[r * LDP + c8];
#pragma unroll
        for (int j = 0; j < 8; ++j) dst[j] = (_Float16)src[j];
      }
    } else {
#pragma unroll
      for (int i = 0; i < 4; ++i) {
        int u = threadIdx.x + 128 * i;
        int r = u >> 3, c8 = (u & 7) * 8;   // r = n, c8 = k chunk base
        const float* src = Bm + (long long)(kk + c8) * ldb + col0 + r;
        _Float16* dst = &Bh[r * LDP + c8];
#pragma unroll
        for (int j = 0; j < 8; ++j) dst[j] = (_Float16)src[(long long)j * ldb];
      }
    }
    __syncthreads();

    // ---- two K=32 WMMA steps from LDS
#pragma unroll
    for (int ks = 0; ks < KT; ks += 32) {
      int ma0 = wm * 32 + lm, ma1 = ma0 + 16;
      int nb0 = wn * 32 + lm, nb1 = nb0 + 16;
      // A fragment: slots 0-7 -> k = ks+8g+0..7 ; slots 8-15 -> k = ks+8g+16..23
      v8h al0 = *(const v8h*)&Ah[ma0 * LDP + ks + 8 * g];
      v8h ah0 = *(const v8h*)&Ah[ma0 * LDP + ks + 8 * g + 16];
      v8h al1 = *(const v8h*)&Ah[ma1 * LDP + ks + 8 * g];
      v8h ah1 = *(const v8h*)&Ah[ma1 * LDP + ks + 8 * g + 16];
      // B fragment: slots 0-7 -> k = ks+8g+0..7 ; slots 8-15 -> k = ks+8g+16..23
      v8h bl0 = *(const v8h*)&Bh[nb0 * LDP + ks + 8 * g];
      v8h bh0 = *(const v8h*)&Bh[nb0 * LDP + ks + 8 * g + 16];
      v8h bl1 = *(const v8h*)&Bh[nb1 * LDP + ks + 8 * g];
      v8h bh1 = *(const v8h*)&Bh[nb1 * LDP + ks + 8 * g + 16];
      v16h a0, a1, b0, b1;
#pragma unroll
      for (int j = 0; j < 8; ++j) {
        a0[j] = al0[j]; a0[8 + j] = ah0[j];
        a1[j] = al1[j]; a1[8 + j] = ah1[j];
        b0[j] = bl0[j]; b0[8 + j] = bh0[j];
        b1[j] = bl1[j]; b1[8 + j] = bh1[j];
      }
      acc00 = __builtin_amdgcn_wmma_f32_16x16x32_f16(false, a0, false, b0, (short)0, acc00, false, false);
      acc01 = __builtin_amdgcn_wmma_f32_16x16x32_f16(false, a0, false, b1, (short)0, acc01, false, false);
      acc10 = __builtin_amdgcn_wmma_f32_16x16x32_f16(false, a1, false, b0, (short)0, acc10, false, false);
      acc11 = __builtin_amdgcn_wmma_f32_16x16x32_f16(false, a1, false, b1, (short)0, acc11, false, false);
    }
    __syncthreads();
  }

  // ---- store (C 16x16 f32 layout: n = lane&15, m = v + 8*(lane>>4))
  int wr0 = row0 + wm * 32, wc0 = col0 + wn * 32;
  int c0 = wc0 + lm, c1 = wc0 + 16 + lm;
  float bb0 = bias ? bias[c0] : 0.f;
  float bb1 = bias ? bias[c1] : 0.f;
#pragma unroll
  for (int v = 0; v < 8; ++v) {
    int mr = v + 8 * g;
    float x00 = acc00[v] * scale + bb0;
    float x01 = acc01[v] * scale + bb1;
    float x10 = acc10[v] * scale + bb0;
    float x11 = acc11[v] * scale + bb1;
    if (ACT) { x00 = gelu_f(x00); x01 = gelu_f(x01); x10 = gelu_f(x10); x11 = gelu_f(x11); }
    C[(long long)(wr0 + mr)      * ldc + c0] = x00;
    C[(long long)(wr0 + mr)      * ldc + c1] = x01;
    C[(long long)(wr0 + 16 + mr) * ldc + c0] = x10;
    C[(long long)(wr0 + 16 + mr) * ldc + c1] = x11;
  }
}

// ---------------------------------------------------------------- embedding
// out[b,l,d] = circ_conv(x)[b,l,d] + pos(l,d) + sum_j table(mark_j, d)
__global__ void embed_kernel(const float* __restrict__ x, const int* __restrict__ mark,
                             const float* __restrict__ convw, float* __restrict__ out, int L)
{
  long long idx = (long long)blockIdx.x * 256 + threadIdx.x;
  long long total = (long long)NB * L * DM;
  if (idx >= total) return;
  int d = (int)(idx % DM);
  long long bl = idx / DM;
  int l = (int)(bl % L);
  int b = (int)(bl / L);

  float acc = 0.f;
#pragma unroll
  for (int t = 0; t < 3; ++t) {
    int ls = l + t - 1; ls = (ls + L) % L;
    const float* xr = x + ((long long)b * L + ls) * 7;
    const float* wr = convw + (long long)d * 21 + t;   // w[d,i,t] at d*21 + i*3 + t
#pragma unroll
    for (int i = 0; i < 7; ++i) acc += xr[i] * wr[i * 3];
  }
  int p = d >> 1;
  float freq = __expf(-(2.0f * p) * (9.210340371976184f / (float)DM));
  float ap = (float)l * freq;
  float pos = (d & 1) ? __cosf(ap) : __sinf(ap);
  const int* mk = mark + ((long long)b * L + l) * 4;
  float temp = 0.f;
#pragma unroll
  for (int j = 0; j < 4; ++j) {
    float a = (float)mk[j] * freq;
    temp += (d & 1) ? __cosf(a) : __sinf(a);
  }
  out[idx] = acc + pos + temp;
}

// ---------------------------------------------------------------- add + layernorm
// y[row,:] = LN(x[row,:] + a[row,:]) * g + b   (a may be null)
__global__ void add_ln_kernel(const float* __restrict__ x, const float* __restrict__ a,
                              const float* __restrict__ g, const float* __restrict__ bt,
                              float* __restrict__ y)
{
  __shared__ float red[256];
  long long row = blockIdx.x;
  int t = threadIdx.x;
  const float* xr = x + row * DM;
  float v0 = xr[t]       + (a ? a[row * DM + t]       : 0.f);
  float v1 = xr[t + 256] + (a ? a[row * DM + t + 256] : 0.f);
  red[t] = v0 + v1; __syncthreads();
  for (int s = 128; s > 0; s >>= 1) { if (t < s) red[t] += red[t + s]; __syncthreads(); }
  float mean = red[0] * (1.0f / DM);
  __syncthreads();
  float d0 = v0 - mean, d1 = v1 - mean;
  red[t] = d0 * d0 + d1 * d1; __syncthreads();
  for (int s = 128; s > 0; s >>= 1) { if (t < s) red[t] += red[t + s]; __syncthreads(); }
  float inv = rsqrtf(red[0] * (1.0f / DM) + LN_EPS);
  y[row * DM + t]       = d0 * inv * g[t]       + bt[t];
  y[row * DM + t + 256] = d1 * inv * g[t + 256] + bt[t + 256];
}

// ---------------------------------------------------------------- prob-attn: sampled sparsity measure
__global__ void qk_sample_M(const float* __restrict__ Q, const float* __restrict__ K,
                            float* __restrict__ Mout, int L, int U, int LK, unsigned seed)
{
  int idx = blockIdx.x * 256 + threadIdx.x;
  if (idx >= NB * NH * L) return;
  int l = idx % L; int bh = idx / L;
  int h = bh % NH; int b = bh / NH;
  const float* q = Q + ((long long)b * L + l) * DM + h * DHD;
  float mx = -1e30f, sm = 0.f;
  for (int u = 0; u < U; ++u) {
    int ks = (int)(wang_hash(seed ^ (unsigned)(l * 131 + u * 7919)) % (unsigned)LK);
    const float* kp = K + ((long long)b * LK + ks) * DM + h * DHD;
    float d = 0.f;
#pragma unroll
    for (int i = 0; i < DHD; ++i) d += q[i] * kp[i];
    mx = fmaxf(mx, d); sm += d;
  }
  Mout[idx] = mx - sm / (float)LK;
}

// ---------------------------------------------------------------- top-k selection per (b,h)
__global__ void topk_kernel(const float* __restrict__ M, int* __restrict__ top, int L, int u)
{
  __shared__ float sv[1024];
  __shared__ float rv[256];
  __shared__ int   ri[256];
  int bh = blockIdx.x;
  const float* m = M + (long long)bh * L;
  for (int i = threadIdx.x; i < L; i += 256) sv[i] = m[i];
  __syncthreads();
  for (int it = 0; it < u; ++it) {
    float best = -1e30f; int bi = 0;
    for (int i = threadIdx.x; i < L; i += 256)
      if (sv[i] > best) { best = sv[i]; bi = i; }
    rv[threadIdx.x] = best; ri[threadIdx.x] = bi;
    __syncthreads();
    for (int s = 128; s > 0; s >>= 1) {
      if (threadIdx.x < s && rv[threadIdx.x + s] > rv[threadIdx.x]) {
        rv[threadIdx.x] = rv[threadIdx.x + s]; ri[threadIdx.x] = ri[threadIdx.x + s];
      }
      __syncthreads();
    }
    if (threadIdx.x == 0) { top[(long long)bh * u + it] = ri[0]; sv[ri[0]] = -1e30f; }
    __syncthreads();
  }
}

// ---------------------------------------------------------------- ctx init: mean / cumsum of V
__global__ void ctx_mean(const float* __restrict__ V, float* __restrict__ ctx, int L)
{
  int idx = blockIdx.x * 256 + threadIdx.x;
  if (idx >= NB * DM) return;
  int c = idx % DM; int b = idx / DM;
  const float* v = V + (long long)b * L * DM + c;
  float s = 0.f;
  for (int l = 0; l < L; ++l) s += v[(long long)l * DM];
  s /= (float)L;
  float* o = ctx + (long long)b * L * DM + c;
  for (int l = 0; l < L; ++l) o[(long long)l * DM] = s;
}

__global__ void ctx_cumsum(const float* __restrict__ V, float* __restrict__ ctx, int L)
{
  int idx = blockIdx.x * 256 + threadIdx.x;
  if (idx >= NB * DM) return;
  int c = idx % DM; int b = idx / DM;
  const float* v = V + (long long)b * L * DM + c;
  float* o = ctx + (long long)b * L * DM + c;
  float s = 0.f;
  for (int l = 0; l < L; ++l) { s += v[(long long)l * DM]; o[(long long)l * DM] = s; }
}

// ---------------------------------------------------------------- prob-attn update of top rows
__global__ void prob_attn(const float* __restrict__ Q, const float* __restrict__ K,
                          const float* __restrict__ V, const int* __restrict__ top,
                          float* __restrict__ ctx, int L, int LK, int u, int mask)
{
  __shared__ float sc[1024];
  __shared__ float red[256];
  int z = blockIdx.x;
  int ui = z % u; int bh = z / u;
  int h = bh % NH; int b = bh / NH;
  int row = top[(long long)bh * u + ui];
  const float* q = Q + ((long long)b * L + row) * DM + h * DHD;

  for (int s = threadIdx.x; s < LK; s += 256) {
    const float* kp = K + ((long long)b * LK + s) * DM + h * DHD;
    float d = 0.f;
#pragma unroll
    for (int i = 0; i < DHD; ++i) d += q[i] * kp[i];
    d *= 0.125f;                       // 1/sqrt(64)
    if (mask && s > row) d = -1e30f;
    sc[s] = d;
  }
  __syncthreads();
  float mx = -1e30f;
  for (int s = threadIdx.x; s < LK; s += 256) mx = fmaxf(mx, sc[s]);
  red[threadIdx.x] = mx; __syncthreads();
  for (int s = 128; s > 0; s >>= 1) { if (threadIdx.x < s) red[threadIdx.x] = fmaxf(red[threadIdx.x], red[threadIdx.x + s]); __syncthreads(); }
  mx = red[0]; __syncthreads();
  float sm = 0.f;
  for (int s = threadIdx.x; s < LK; s += 256) { float e = __expf(sc[s] - mx); sc[s] = e; sm += e; }
  red[threadIdx.x] = sm; __syncthreads();
  for (int s = 128; s > 0; s >>= 1) { if (threadIdx.x < s) red[threadIdx.x] += red[threadIdx.x + s]; __syncthreads(); }
  float inv = 1.0f / red[0];
  if (threadIdx.x < DHD) {
    int d = threadIdx.x;
    const float* vp = V + (long long)b * LK * DM + h * DHD + d;
    float o = 0.f;
    for (int s = 0; s < LK; ++s) o += sc[s] * vp[(long long)s * DM];
    ctx[((long long)b * L + row) * DM + h * DHD + d] = o * inv;
  }
}

// ---------------------------------------------------------------- row softmax, width 256 (cross attention)
__global__ void softmax256(float* __restrict__ S)
{
  __shared__ float red[256];
  long long row = blockIdx.x;
  int t = threadIdx.x;
  float v = S[row * 256 + t];
  red[t] = v; __syncthreads();
  for (int s = 128; s > 0; s >>= 1) { if (t < s) red[t] = fmaxf(red[t], red[t + s]); __syncthreads(); }
  float mx = red[0]; __syncthreads();
  float e = __expf(v - mx);
  red[t] = e; __syncthreads();
  for (int s = 128; s > 0; s >>= 1) { if (t < s) red[t] += red[t + s]; __syncthreads(); }
  S[row * 256 + t] = e / red[0];
}

// ---------------------------------------------------------------- distil: im2col for circular conv k=3
__global__ void im2col_kernel(const float* __restrict__ x, float* __restrict__ out, int L)
{
  long long idx = (long long)blockIdx.x * 256 + threadIdx.x;
  long long total = (long long)NB * L * 1536;
  if (idx >= total) return;
  int col = (int)(idx % 1536);
  long long bl = idx / 1536;
  int l = (int)(bl % L); int b = (int)(bl / L);
  int t = col / DM, i = col % DM;
  int ls = (l + t - 1 + L) % L;
  out[idx] = x[((long long)b * L + ls) * DM + i];
}

// wp[d, t*512+i] = w[d,i,t]  (w is (512,512,3))
__global__ void repack_wd(const float* __restrict__ w, float* __restrict__ wp)
{
  int idx = blockIdx.x * 256 + threadIdx.x;
  if (idx >= DM * 1536) return;
  int col = idx % 1536; int d = idx / 1536;
  int t = col / DM, i = col % DM;
  wp[idx] = w[((long long)d * DM + i) * 3 + t];
}

__global__ void bn_stats(const float* __restrict__ y, float* __restrict__ mean,
                         float* __restrict__ var, long long rows)
{
  __shared__ float r1[256], r2[256];
  int d = blockIdx.x;
  float s = 0.f, s2 = 0.f;
  for (long long r = threadIdx.x; r < rows; r += 256) {
    float v = y[r * DM + d]; s += v; s2 += v * v;
  }
  r1[threadIdx.x] = s; r2[threadIdx.x] = s2; __syncthreads();
  for (int st = 128; st > 0; st >>= 1) {
    if (threadIdx.x < st) { r1[threadIdx.x] += r1[threadIdx.x + st]; r2[threadIdx.x] += r2[threadIdx.x + st]; }
    __syncthreads();
  }
  if (threadIdx.x == 0) {
    float m = r1[0] / (float)rows;
    mean[d] = m; var[d] = r2[0] / (float)rows - m * m;
  }
}

__global__ void elu_maxpool(const float* __restrict__ y, const float* __restrict__ mean,
                            const float* __restrict__ var, float* __restrict__ out, int L)
{
  long long idx = (long long)blockIdx.x * 256 + threadIdx.x;
  int Lo = L >> 1;
  long long total = (long long)NB * Lo * DM;
  if (idx >= total) return;
  int d = (int)(idx % DM);
  long long bl = idx / DM;
  int lo = (int)(bl % Lo); int b = (int)(bl / Lo);
  float m = mean[d];
  float iv = rsqrtf(var[d] + LN_EPS);
  float best = -1e30f;
#pragma unroll
  for (int t = -1; t <= 1; ++t) {
    int l = 2 * lo + t;
    if (l < 0 || l >= L) continue;
    float v = (y[((long long)b * L + l) * DM + d] - m) * iv;
    v = v > 0.f ? v : (__expf(v) - 1.0f);
    best = fmaxf(best, v);
  }
  out[idx] = best;
}

// ---------------------------------------------------------------- final projection 512->7, last 512 rows
__global__ void proj_kernel(const float* __restrict__ x, const float* __restrict__ w,
                            const float* __restrict__ bias, float* __restrict__ out)
{
  int idx = blockIdx.x * 256 + threadIdx.x;
  if (idx >= NB * 512 * 7) return;
  int c = idx % 7; int t = (idx / 7) % 512; int b = idx / (7 * 512);
  const float* xr = x + ((long long)b * 1024 + 512 + t) * DM;
  const float* wr = w + (long long)c * DM;
  float s = bias[c];
  for (int i = 0; i < DM; ++i) s += xr[i] * wr[i];
  out[idx] = s;
}

// ---------------------------------------------------------------- host side
static inline dim3 g256(long long n) { return dim3((unsigned)((n + 255) / 256)); }

static inline void run_gemm(hipStream_t st, int transB, int act,
                            const float* A, int lda, long long sAb, long long sAh,
                            const float* Bm, int ldb, long long sBb, long long sBh,
                            float* C, int ldc, long long sCb, long long sCh,
                            const float* bias, int M, int N, int K, int Hd, int Z, float scale)
{
  dim3 g(N / 64, M / 64, Z), b(128);
  if (transB) {
    if (act) wmma_gemm<1,1><<<g,b,0,st>>>(A,lda,sAb,sAh,Bm,ldb,sBb,sBh,C,ldc,sCb,sCh,bias,M,N,K,Hd,scale);
    else     wmma_gemm<1,0><<<g,b,0,st>>>(A,lda,sAb,sAh,Bm,ldb,sBb,sBh,C,ldc,sCb,sCh,bias,M,N,K,Hd,scale);
  } else {
    if (act) wmma_gemm<0,1><<<g,b,0,st>>>(A,lda,sAb,sAh,Bm,ldb,sBb,sBh,C,ldc,sCb,sCh,bias,M,N,K,Hd,scale);
    else     wmma_gemm<0,0><<<g,b,0,st>>>(A,lda,sAb,sAh,Bm,ldb,sBb,sBh,C,ldc,sCb,sCh,bias,M,N,K,Hd,scale);
  }
}

static inline int ufac(int n) {
  int v = (int)(5.0 * ceil(log((double)n)));
  return v < n ? v : n;
}

extern "C" void kernel_launch(void* const* d_in, const int* in_sizes, int n_in,
                              void* d_out, int out_size, void* d_ws, size_t ws_size,
                              hipStream_t stream)
{
  (void)in_sizes; (void)n_in; (void)out_size; (void)ws_size;
  auto F = [&](int i) { return (const float*)d_in[i]; };

  // pytree-flat (alphabetical) parameter indices
  // 0 x_enc, 1 x_mark_enc, 2 x_dec, 3 x_mark_dec
  // 4 dec_conv_w; dec_layers l: base 5+26*l
  //   +0 b1 +1 b2 | cross: +2 kb +3 kw +4 ob +5 ow +6 qb +7 qw +8 vb +9 vw
  //   +10 ln1b +11 ln1g +12 ln2b +13 ln2g +14 ln3b +15 ln3g
  //   | self: +16 kb +17 kw +18 ob +19 ow +20 qb +21 qw +22 vb +23 vw | +24 w1 +25 w2
  // 57 dec_norm_b, 58 dec_norm_g
  // 59 distil0.b, 60 distil0.w, 61 distil1.b, 62 distil1.w
  // 63 enc_conv_w; enc_layers l: base 64+16*l
  //   +0 b1 +1 b2 +2 kb +3 kw +4 ln1b +5 ln1g +6 ln2b +7 ln2g
  //   +8 ob +9 ow +10 qb +11 qw +12 vb +13 vw +14 w1 +15 w2
  // 112 enc_norm_b, 113 enc_norm_g, 114 proj_b, 115 proj_w

  float* W = (float*)d_ws;
  const long long SZ4M = 4194304LL;      // NB*1024*512
  float* X   = W;
  float* X2  = W + 1 * SZ4M;
  float* Qb  = W + 2 * SZ4M;
  float* Kb  = W + 3 * SZ4M;
  float* Vb  = W + 4 * SZ4M;
  float* CTX = W + 5 * SZ4M;
  float* T1  = W + 6 * SZ4M;
  float* FF  = W + 7 * SZ4M;             // 16,777,216 floats (FFN mid / im2col / scores)
  float* ENC = FF + 16777216;            // 1,048,576
  float* Mb  = ENC + 1048576;            // 65,536
  float* ST  = Mb + 65536;               // 1024 (mean|var)
  float* WD  = ST + 1024;                // 786,432
  int*  TOP  = (int*)(WD + 786432);      // 4096 ints

  // =============================== ENCODER ===============================
  embed_kernel<<<g256((long long)NB*1024*DM),256,0,stream>>>(F(0), (const int*)d_in[1], F(63), X, 1024);

  float* cur = X; float* alt = X2;
  int L = 1024;
  for (int l = 0; l < 3; ++l) {
    int bs = 64 + 16 * l;
    const float *b1=F(bs+0), *b2=F(bs+1), *kb=F(bs+2), *kw=F(bs+3);
    const float *ln1b=F(bs+4), *ln1g=F(bs+5), *ln2b=F(bs+6), *ln2g=F(bs+7);
    const float *ob=F(bs+8), *ow=F(bs+9), *qb=F(bs+10), *qw=F(bs+11);
    const float *vb=F(bs+12), *vw=F(bs+13), *w1=F(bs+14), *w2=F(bs+15);
    int M = NB * L;
    run_gemm(stream,1,0, cur,DM,0,0, qw,DM,0,0, Qb,DM,0,0, qb, M,DM,DM, 1,1, 1.0f);
    run_gemm(stream,1,0, cur,DM,0,0, kw,DM,0,0, Kb,DM,0,0, kb, M,DM,DM, 1,1, 1.0f);
    run_gemm(stream,1,0, cur,DM,0,0, vw,DM,0,0, Vb,DM,0,0, vb, M,DM,DM, 1,1, 1.0f);
    int u = ufac(L);
    qk_sample_M<<<g256((long long)NB*NH*L),256,0,stream>>>(Qb, Kb, Mb, L, u, L, 1234u + l);
    topk_kernel<<<NB*NH,256,0,stream>>>(Mb, TOP, L, u);
    ctx_mean<<<g256(NB*DM),256,0,stream>>>(Vb, CTX, L);
    prob_attn<<<NB*NH*u,256,0,stream>>>(Qb, Kb, Vb, TOP, CTX, L, L, u, 0);
    run_gemm(stream,1,0, CTX,DM,0,0, ow,DM,0,0, T1,DM,0,0, ob, M,DM,DM, 1,1, 1.0f);
    add_ln_kernel<<<M,256,0,stream>>>(cur, T1, ln1g, ln1b, alt);
    { float* t = cur; cur = alt; alt = t; }
    run_gemm(stream,1,1, cur,DM,0,0, w1,DM,0,0, FF,DFF,0,0, b1, M,DFF,DM, 1,1, 1.0f);
    run_gemm(stream,1,0, FF,DFF,0,0, w2,DFF,0,0, T1,DM,0,0, b2, M,DM,DFF, 1,1, 1.0f);
    add_ln_kernel<<<M,256,0,stream>>>(cur, T1, ln2g, ln2b, alt);
    { float* t = cur; cur = alt; alt = t; }
    if (l < 2) {
      const float* db = F(59 + 2 * l);
      const float* dw = F(60 + 2 * l);
      im2col_kernel<<<g256((long long)M*1536),256,0,stream>>>(cur, FF, L);
      repack_wd<<<g256(DM*1536),256,0,stream>>>(dw, WD);
      run_gemm(stream,1,0, FF,1536,0,0, WD,1536,0,0, T1,DM,0,0, db, M,DM,1536, 1,1, 1.0f);
      bn_stats<<<DM,256,0,stream>>>(T1, ST, ST + 512, (long long)M);
      elu_maxpool<<<g256((long long)NB*(L/2)*DM),256,0,stream>>>(T1, ST, ST + 512, alt, L);
      { float* t = cur; cur = alt; alt = t; }
      L >>= 1;
    }
  }
  add_ln_kernel<<<NB*L,256,0,stream>>>(cur, nullptr, F(113), F(112), ENC);   // final enc LN -> (8,256,512)

  // =============================== DECODER ===============================
  cur = X; alt = X2;
  embed_kernel<<<g256((long long)NB*1024*DM),256,0,stream>>>(F(2), (const int*)d_in[3], F(4), cur, 1024);

  const int LD = 1024, S = 256;
  const int MD = NB * LD, MS = NB * S;
  for (int l = 0; l < 2; ++l) {
    int bs = 5 + 26 * l;
    const float *b1=F(bs+0), *b2=F(bs+1);
    const float *ckb=F(bs+2), *ckw=F(bs+3), *cob=F(bs+4), *cow=F(bs+5);
    const float *cqb=F(bs+6), *cqw=F(bs+7), *cvb=F(bs+8), *cvw=F(bs+9);
    const float *ln1b=F(bs+10), *ln1g=F(bs+11), *ln2b=F(bs+12), *ln2g=F(bs+13);
    const float *ln3b=F(bs+14), *ln3g=F(bs+15);
    const float *skb=F(bs+16), *skw=F(bs+17), *sob=F(bs+18), *sow=F(bs+19);
    const float *sqb=F(bs+20), *sqw=F(bs+21), *svb=F(bs+22), *svw=F(bs+23);
    const float *w1=F(bs+24), *w2=F(bs+25);

    // ---- masked self prob-attention
    run_gemm(stream,1,0, cur,DM,0,0, sqw,DM,0,0, Qb,DM,0,0, sqb, MD,DM,DM, 1,1, 1.0f);
    run_gemm(stream,1,0, cur,DM,0,0, skw,DM,0,0, Kb,DM,0,0, skb, MD,DM,DM, 1,1, 1.0f);
    run_gemm(stream,1,0, cur,DM,0,0, svw,DM,0,0, Vb,DM,0,0, svb, MD,DM,DM, 1,1, 1.0f);
    int u = ufac(LD);
    qk_sample_M<<<g256((long long)NB*NH*LD),256,0,stream>>>(Qb, Kb, Mb, LD, u, LD, 5678u + l);
    topk_kernel<<<NB*NH,256,0,stream>>>(Mb, TOP, LD, u);
    ctx_cumsum<<<g256(NB*DM),256,0,stream>>>(Vb, CTX, LD);
    prob_attn<<<NB*NH*u,256,0,stream>>>(Qb, Kb, Vb, TOP, CTX, LD, LD, u, 1);
    run_gemm(stream,1,0, CTX,DM,0,0, sow,DM,0,0, T1,DM,0,0, sob, MD,DM,DM, 1,1, 1.0f);
    add_ln_kernel<<<MD,256,0,stream>>>(cur, T1, ln1g, ln1b, alt);
    { float* t = cur; cur = alt; alt = t; }

    // ---- cross full attention vs ENC (S=256)
    run_gemm(stream,1,0, cur,DM,0,0, cqw,DM,0,0, Qb,DM,0,0, cqb, MD,DM,DM, 1,1, 1.0f);
    run_gemm(stream,1,0, ENC,DM,0,0, ckw,DM,0,0, Kb,DM,0,0, ckb, MS,DM,DM, 1,1, 1.0f);
    run_gemm(stream,1,0, ENC,DM,0,0, cvw,DM,0,0, Vb,DM,0,0, cvb, MS,DM,DM, 1,1, 1.0f);
    // scores[b,h] = Q_bh (1024x64) * K_bh^T (256x64)^T / 8
    run_gemm(stream,1,0, Qb,DM,(long long)LD*DM,DHD,
             Kb,DM,(long long)S*DM,DHD,
             FF,S,(long long)NH*LD*S,(long long)LD*S,
             nullptr, LD,S,DHD, NH, NB*NH, 0.125f);
    softmax256<<<NB*NH*LD,256,0,stream>>>(FF);
    // ctx[b,h] = attn (1024x256) * V_bh (256x64)
    run_gemm(stream,0,0, FF,S,(long long)NH*LD*S,(long long)LD*S,
             Vb,DM,(long long)S*DM,DHD,
             CTX,DM,(long long)LD*DM,DHD,
             nullptr, LD,DHD,S, NH, NB*NH, 1.0f);
    run_gemm(stream,1,0, CTX,DM,0,0, cow,DM,0,0, T1,DM,0,0, cob, MD,DM,DM, 1,1, 1.0f);
    add_ln_kernel<<<MD,256,0,stream>>>(cur, T1, ln2g, ln2b, alt);
    { float* t = cur; cur = alt; alt = t; }

    // ---- FFN
    run_gemm(stream,1,1, cur,DM,0,0, w1,DM,0,0, FF,DFF,0,0, b1, MD,DFF,DM, 1,1, 1.0f);
    run_gemm(stream,1,0, FF,DFF,0,0, w2,DFF,0,0, T1,DM,0,0, b2, MD,DM,DFF, 1,1, 1.0f);
    add_ln_kernel<<<MD,256,0,stream>>>(cur, T1, ln3g, ln3b, alt);
    { float* t = cur; cur = alt; alt = t; }
  }
  add_ln_kernel<<<MD,256,0,stream>>>(cur, nullptr, F(58), F(57), alt);       // final dec LN
  proj_kernel<<<g256(NB*512*7),256,0,stream>>>(alt, F(115), F(114), (float*)d_out);
}
